// ChordTransformer_37520834298166
// MI455X (gfx1250) — matrix operand order repaired and checked by reading, not compiled, not run on hardware
//
#include <hip/hip_runtime.h>
#include <hip/hip_bf16.h>

typedef __attribute__((ext_vector_type(16))) _Float16 v16h;
typedef __attribute__((ext_vector_type(8)))  float    v8f;

// Model constants
constexpr int Bn  = 64;
constexpr int Tn  = 1024;
constexpr int Dn  = 64;
constexpr int Hn  = 2;
constexpr int HDn = 32;
constexpr int FFn = 128;
constexpr int Cn  = 128;
constexpr int Nn  = Bn * Tn;        // 65536 tokens
constexpr int NDn = Nn * Dn;        // 4194304 floats per activation tensor

// ---------------------------------------------------------------------------
// WMMA operand index helpers (wave32, v_wmma_f32_16x16x32_f16)
// A (16x32, MxK): lane L holds row L%16; half = L/16.
//   half 0: j<8 -> K=j,   j>=8 -> K=8+j   (K 0..7, 16..23)
//   half 1: j<8 -> K=8+j, j>=8 -> K=16+j  (K 8..15, 24..31)
__device__ __forceinline__ int a_kmap(int j, int half) {
    return half * 8 + j + ((j >= 8) ? 8 : 0);
}
// B (32x16, KxN): lane L holds column L%16; K mapping same as A.
__device__ __forceinline__ int b_kmap(int j, int half) {
    return half * 8 + j + ((j >= 8) ? 8 : 0);
}
// C/D (16x16 f32): lane L<16 -> col L, rows 0..7 in regs; lane>=16 -> col
// L-16, rows 8..15.

// 16-lane max butterfly with DPP (stays within a DPP row of 16 lanes):
// quad_perm xor1 (0xB1), quad_perm xor2 (0x4E), ROW_HALF_MIRROR (0x141),
// ROW_MIRROR (0x140). Pure VALU - no LDS traffic.
__device__ __forceinline__ float row16_max(float v) {
    int x, y;
    x = __float_as_int(v);
    y = __builtin_amdgcn_update_dpp(x, x, 0xB1, 0xF, 0xF, false);
    v = fmaxf(v, __int_as_float(y));
    x = __float_as_int(v);
    y = __builtin_amdgcn_update_dpp(x, x, 0x4E, 0xF, 0xF, false);
    v = fmaxf(v, __int_as_float(y));
    x = __float_as_int(v);
    y = __builtin_amdgcn_update_dpp(x, x, 0x141, 0xF, 0xF, false);
    v = fmaxf(v, __int_as_float(y));
    x = __float_as_int(v);
    y = __builtin_amdgcn_update_dpp(x, x, 0x140, 0xF, 0xF, false);
    v = fmaxf(v, __int_as_float(y));
    return v;
}

// ---------------------------------------------------------------------------
// Encoder input: x = (melody @ Win + bin) * sqrt(D) + PE(t)
__global__ void enc_in_kernel(const float* __restrict__ mel,
                              const float* __restrict__ W,
                              const float* __restrict__ b,
                              float* __restrict__ out) {
    int idx = blockIdx.x * blockDim.x + threadIdx.x;   // over N*D
    int n = idx >> 6;
    int d = idx & 63;
    int t = n & (Tn - 1);
    float acc = b[d];
#pragma unroll
    for (int f = 0; f < 5; ++f) acc += mel[n * 5 + f] * W[f * Dn + d];
    int i2 = d & ~1;  // 2*i
    float w = __expf(-(float)i2 * (9.210340371976184f / 64.0f));  // 10000^(-2i/64)
    float ang = (float)t * w;
    float pe = (d & 1) ? __cosf(ang) : __sinf(ang);
    out[idx] = acc * 8.0f + pe;   // sqrt(64) = 8
}

// ---------------------------------------------------------------------------
// Decoder input: y = emb[token]*sqrt(D) + PE(0)   (PE(0) = [0,1,0,1,...])
__global__ void embed_kernel(const int* __restrict__ chords,
                             const float* __restrict__ emb,
                             float* __restrict__ y) {
    int idx = blockIdx.x * blockDim.x + threadIdx.x;   // over N*D
    int n = idx >> 6;
    int d = idx & 63;
    int t = n & (Tn - 1);
    int b = n >> 10;
    int tok = (t == 0) ? 0 : chords[b * Tn + t - 1];
    float pe0 = (d & 1) ? 1.0f : 0.0f;
    y[idx] = emb[tok * Dn + d] * 8.0f + pe0;
}

// ---------------------------------------------------------------------------
// GEMM: out[N x M] = A[N x K] @ W[K x M] + bias (+ReLU).
// Block = 128 threads = 4 waves; wave w owns a 16x16 tile at cols
// blockIdx.y*64 + w*16, rows blockIdx.x*16.
template <bool RELU>
__global__ void gemm_bias_kernel(const float* __restrict__ A,
                                 const float* __restrict__ W,
                                 const float* __restrict__ bias,
                                 float* __restrict__ out,
                                 int K, int M) {
    const int lane = threadIdx.x & 31;
    const int wave = threadIdx.x >> 5;
    const int half = lane >> 4;
    const int lm   = lane & 15;
    const int row0 = blockIdx.x * 16;
    const int c0   = blockIdx.y * 64 + wave * 16;
    const int arow = row0 + lm;
    const int col  = c0 + lm;

    v8f acc = {};
    for (int k0 = 0; k0 < K; k0 += 32) {
        v16h a, bm;
#pragma unroll
        for (int j = 0; j < 16; ++j) {
            a[j]  = (_Float16)A[arow * K + k0 + a_kmap(j, half)];
            bm[j] = (_Float16)W[(k0 + b_kmap(j, half)) * M + col];
        }
        acc = __builtin_amdgcn_wmma_f32_16x16x32_f16(false, a, false, bm,
                                                     (short)0, acc, false, false);
    }
#pragma unroll
    for (int r = 0; r < 8; ++r) {
        int m = half * 8 + r;
        float v = acc[r] + bias[col];
        if (RELU) v = fmaxf(v, 0.0f);
        out[(row0 + m) * M + col] = v;
    }
}

// ---------------------------------------------------------------------------
// Flash attention (no mask). Q,K,V,O all (B,T,H,HD) packed as (N x 64) with
// head h occupying columns [h*32, h*32+32). One wave per (b, h, q-tile of 16).
// Row max: DPP butterfly. Row sum: extra WMMA against an all-ones B operand
// (accumulates running row sums directly in C-layout registers).
__global__ void attn_kernel(const float* __restrict__ Q,
                            const float* __restrict__ Km,
                            const float* __restrict__ Vm,
                            float* __restrict__ O) {
    __shared__ _Float16 Psh[4][16 * 32];   // per-wave probability tile
    const int lane = threadIdx.x & 31;
    const int wave = threadIdx.x >> 5;
    const int half = lane >> 4;
    const int lm   = lane & 15;

    int gw = blockIdx.x * 4 + wave;       // 0 .. B*H*(T/16)-1 = 8191
    const int qt = gw & 63;               // q tile (T/16 = 64)
    const int h  = (gw >> 6) & 1;
    const int b  = gw >> 7;
    const float scale = 0.17677669529663689f;  // 1/sqrt(HD)

    // Q as A-operand, pre-scaled by 1/sqrt(HD). Loaded once (K-dim = HD = 32).
    const int qrow = (b * Tn + qt * 16 + lm) * Dn + h * HDn;
    v16h aq;
#pragma unroll
    for (int j = 0; j < 16; ++j)
        aq[j] = (_Float16)(Q[qrow + a_kmap(j, half)] * scale);

    // Constant all-ones B operand for row-sum WMMA.
    v16h bone;
#pragma unroll
    for (int j = 0; j < 16; ++j) bone[j] = (_Float16)1.0f;

    v8f o0 = {}, o1 = {};                 // hd cols [0,16) and [16,32)
    v8f lacc = {};                        // running row sums (all cols equal)
    float mrow[8];                        // running max for rows half*8 + r
#pragma unroll
    for (int r = 0; r < 8; ++r) mrow[r] = -1e30f;

    _Float16* P = Psh[wave];
    const int kvbase = b * Tn * Dn + h * HDn;

    for (int kb = 0; kb < Tn; kb += 32) {
        // Prefetch next chunk's K/V rows (global_prefetch_b8); K/V are
        // re-streamed T/16 times per (b,h), keep them warm in WGP$/L2.
        if (kb + 32 < Tn) {
            __builtin_prefetch(Km + kvbase + (kb + 32 + lane) * Dn, 0, 1);
            __builtin_prefetch(Vm + kvbase + (kb + 32 + lane) * Dn, 0, 1);
        }
        // Score tiles: S0 = keys kb..kb+15, S1 = keys kb+16..kb+31
        v16h bk0, bk1;
#pragma unroll
        for (int j = 0; j < 16; ++j) {
            int hd = b_kmap(j, half);
            bk0[j] = (_Float16)Km[kvbase + (kb + lm) * Dn + hd];
            bk1[j] = (_Float16)Km[kvbase + (kb + 16 + lm) * Dn + hd];
        }
        v8f z = {};
        v8f s0 = __builtin_amdgcn_wmma_f32_16x16x32_f16(false, aq, false, bk0,
                                                        (short)0, z, false, false);
        v8f s1 = __builtin_amdgcn_wmma_f32_16x16x32_f16(false, aq, false, bk1,
                                                        (short)0, z, false, false);
        // Online softmax update (row = half*8+r, col = lane%16)
#pragma unroll
        for (int r = 0; r < 8; ++r) {
            float mx = row16_max(fmaxf(s0[r], s1[r]));
            float mnew = fmaxf(mrow[r], mx);
            float corr = __expf(mrow[r] - mnew);
            float p0 = __expf(s0[r] - mnew);
            float p1 = __expf(s1[r] - mnew);
            mrow[r] = mnew;
            o0[r]   *= corr;
            o1[r]   *= corr;
            lacc[r] *= corr;
            int pr = half * 8 + r;
            P[pr * 32 + lm]      = (_Float16)p0;
            P[pr * 32 + 16 + lm] = (_Float16)p1;
        }
        asm volatile("s_wait_dscnt 0x0" ::: "memory");  // wave's LDS stores visible

        // P (16x32) as A-operand
        v16h ap;
#pragma unroll
        for (int j = 0; j < 16; ++j)
            ap[j] = P[lm * 32 + a_kmap(j, half)];

        // V (32 keys x 32 hd) as two B-operands
        v16h bv0, bv1;
#pragma unroll
        for (int j = 0; j < 16; ++j) {
            int kk = b_kmap(j, half);     // key offset within chunk
            const float* vrow = Vm + kvbase + (kb + kk) * Dn;
            bv0[j] = (_Float16)vrow[lm];
            bv1[j] = (_Float16)vrow[16 + lm];
        }
        o0 = __builtin_amdgcn_wmma_f32_16x16x32_f16(false, ap, false, bv0,
                                                    (short)0, o0, false, false);
        o1 = __builtin_amdgcn_wmma_f32_16x16x32_f16(false, ap, false, bv1,
                                                    (short)0, o1, false, false);
        // Row sums: lacc += P @ ones  (every column of the tile gets the sum)
        lacc = __builtin_amdgcn_wmma_f32_16x16x32_f16(false, ap, false, bone,
                                                      (short)0, lacc, false, false);
    }

    const int obase = b * Tn * Dn + h * HDn;
#pragma unroll
    for (int r = 0; r < 8; ++r) {
        int m = half * 8 + r;
        float inv = 1.0f / lacc[r];
        O[obase + (qt * 16 + m) * Dn + lm]      = o0[r] * inv;
        O[obase + (qt * 16 + m) * Dn + 16 + lm] = o1[r] * inv;
    }
}

// ---------------------------------------------------------------------------
// out = LayerNorm(X + R) * g + beta   (row width D=64, one wave per row)
__global__ void add_ln_kernel(const float* __restrict__ X,
                              const float* __restrict__ R,
                              const float* __restrict__ g,
                              const float* __restrict__ be,
                              float* __restrict__ out) {
    int lane = threadIdx.x & 31;
    int row  = (blockIdx.x * blockDim.x + threadIdx.x) >> 5;
    const float* xr = X + row * Dn;
    const float* rr = R + row * Dn;
    float x0 = xr[lane] + rr[lane];
    float x1 = xr[lane + 32] + rr[lane + 32];
    float s = x0 + x1;
    for (int off = 1; off < 32; off <<= 1) s += __shfl_xor(s, off, 32);
    float mean = s * (1.0f / 64.0f);
    float d0 = x0 - mean, d1 = x1 - mean;
    float v = d0 * d0 + d1 * d1;
    for (int off = 1; off < 32; off <<= 1) v += __shfl_xor(v, off, 32);
    float rstd = rsqrtf(v * (1.0f / 64.0f) + 1e-5f);
    out[row * Dn + lane]      = d0 * rstd * g[lane] + be[lane];
    out[row * Dn + lane + 32] = d1 * rstd * g[lane + 32] + be[lane + 32];
}

// ---------------------------------------------------------------------------
extern "C" void kernel_launch(void* const* d_in, const int* in_sizes, int n_in,
                              void* d_out, int out_size, void* d_ws, size_t ws_size,
                              hipStream_t stream) {
    // Input leaves, setup_inputs() insertion order:
    // 0 melody, 1 chords, 2 enc_in.W, 3 enc_in.b,
    // enc[l] base e=4+16l: sa.qW,qb,kW,kb,vW,vb,oW,ob, ln1.g,b, ff1.W,b, ff2.W,b, ln2.g,b
    // 36 emb,
    // dec[l] base d=37+26l: sa.qW..ob (0..7), ln1 (8,9), ca.qW..ob (10..17),
    //                       ln2 (18,19), ff1 (20,21), ff2 (22,23), ln3 (24,25)
    // 89 out.W, 90 out.b
    const float* mel    = (const float*)d_in[0];
    const int*   chords = (const int*)d_in[1];
    auto F32 = [&](int i) { return (const float*)d_in[i]; };

    float* ws   = (float*)d_ws;
    float* xbuf = ws;                 // encoder activations / final enc output
    float* qbuf = ws + 1 * (size_t)NDn;
    float* kbuf = ws + 2 * (size_t)NDn;
    float* vbuf = ws + 3 * (size_t)NDn;
    float* t0   = ws + 4 * (size_t)NDn;
    float* t1   = ws + 5 * (size_t)NDn;
    float* ffb  = ws + 6 * (size_t)NDn;   // N x FF (2*NDn floats)
    float* ybuf = ws + 8 * (size_t)NDn;

    auto gemm = [&](const float* A, int wi, int bi, float* out_, int K, int M,
                    bool relu) {
        dim3 grid(Nn / 16, M / 64);
        if (relu)
            gemm_bias_kernel<true><<<grid, 128, 0, stream>>>(A, F32(wi), F32(bi), out_, K, M);
        else
            gemm_bias_kernel<false><<<grid, 128, 0, stream>>>(A, F32(wi), F32(bi), out_, K, M);
    };
    auto ln = [&](const float* X, const float* R, int gi, int bi, float* out_) {
        add_ln_kernel<<<Nn / 8, 256, 0, stream>>>(X, R, F32(gi), F32(bi), out_);
    };
    const int attnBlocks = (Bn * Hn * (Tn / 16)) / 4;   // 2048

    // ---------------- Encoder ----------------
    enc_in_kernel<<<NDn / 256, 256, 0, stream>>>(mel, F32(2), F32(3), xbuf);
    for (int l = 0; l < 2; ++l) {
        int e = 4 + l * 16;
        gemm(xbuf, e + 0, e + 1, qbuf, Dn, Dn, false);   // q proj
        gemm(xbuf, e + 2, e + 3, kbuf, Dn, Dn, false);   // k proj
        gemm(xbuf, e + 4, e + 5, vbuf, Dn, Dn, false);   // v proj
        attn_kernel<<<attnBlocks, 128, 0, stream>>>(qbuf, kbuf, vbuf, t0);
        gemm(t0, e + 6, e + 7, t1, Dn, Dn, false);       // o proj
        ln(xbuf, t1, e + 8, e + 9, xbuf);                // ln1
        gemm(xbuf, e + 10, e + 11, ffb, Dn, FFn, true);  // ff1 + relu
        gemm(ffb, e + 12, e + 13, t1, FFn, Dn, false);   // ff2
        ln(xbuf, t1, e + 14, e + 15, xbuf);              // ln2
    }
    // xbuf now holds encoder output `enc`

    // ---------------- Decoder (vectorized over T) ----------------
    embed_kernel<<<NDn / 256, 256, 0, stream>>>(chords, F32(36), ybuf);
    for (int l = 0; l < 2; ++l) {
        int d = 37 + l * 26;
        // self-attn over length-1 seq == o_proj(v_proj(y))
        gemm(ybuf, d + 4, d + 5, t0, Dn, Dn, false);     // sa.v
        gemm(t0, d + 6, d + 7, t1, Dn, Dn, false);       // sa.o
        ln(ybuf, t1, d + 8, d + 9, ybuf);                // ln1
        // cross-attention against enc
        gemm(ybuf, d + 10, d + 11, qbuf, Dn, Dn, false); // ca.q from y
        gemm(xbuf, d + 12, d + 13, kbuf, Dn, Dn, false); // ca.k from enc
        gemm(xbuf, d + 14, d + 15, vbuf, Dn, Dn, false); // ca.v from enc
        attn_kernel<<<attnBlocks, 128, 0, stream>>>(qbuf, kbuf, vbuf, t0);
        gemm(t0, d + 16, d + 17, t1, Dn, Dn, false);     // ca.o
        ln(ybuf, t1, d + 18, d + 19, ybuf);              // ln2
        gemm(ybuf, d + 20, d + 21, ffb, Dn, FFn, true);  // ff1 + relu
        gemm(ffb, d + 22, d + 23, t1, FFn, Dn, false);   // ff2
        ln(ybuf, t1, d + 24, d + 25, ybuf);              // ln3
    }

    // ---------------- Output projection -> logits (B,T,C) ----------------
    gemm(ybuf, 89, 90, (float*)d_out, Dn, Cn, false);
}
